// MultiHeadAttention_14508399526494
// MI455X (gfx1250) — compile-verified, matrix operand order checked by reading, not run on hardware
//
#include <hip/hip_runtime.h>

// ---------------------------------------------------------------------------
// MI455X / gfx1250 fused MHA: f16 WMMA GEMMs (f32 accumulate), flash-attention
// with TDM + async global->LDS staging, residual + layernorm.  wave32.
// ---------------------------------------------------------------------------

typedef _Float16     v16h  __attribute__((ext_vector_type(16)));
typedef float        v8f   __attribute__((ext_vector_type(8)));
typedef unsigned int u32x4 __attribute__((ext_vector_type(4)));
typedef int          i32x8 __attribute__((ext_vector_type(8)));
typedef int          i32x4 __attribute__((ext_vector_type(4)));

constexpr int BATCH = 2;
constexpr int SEQ   = 2048;
constexpr int EMB   = 1024;
constexpr int ATT   = 1024;
constexpr int HEADS = 16;
constexpr int HD    = 64;
constexpr int MTOT  = BATCH * SEQ;      // 4096 rows
constexpr float NEGF = -4.2949673e9f;   // -2^32 + 1 (reference mask fill)

// Assemble one 16x32 f16 A/B fragment from LDS laid out [row][K].
// Per ISA §7.12.2 (16-bit, wave32):
//   lanes 0-15 : elements 0..7 = K 0..7,   8..15 = K 16..23
//   lanes 16-31: elements 0..7 = K 8..15,  8..15 = K 24..31
// Caller passes p = &lds[row][kchunk*32 + (lane>>4)*8].
__device__ inline v16h frag_ld(const _Float16* __restrict__ p) {
  v16h f;
#pragma unroll
  for (int i = 0; i < 8; ++i) { f[i] = p[i]; f[i + 8] = p[i + 16]; }
  return f;
}

__device__ inline v8f wmma_f16(v16h a, v16h b, v8f c) {
  return __builtin_amdgcn_wmma_f32_16x16x32_f16(
      false, a, false, b, (short)0, c, false, false);
}

// ---------------------------------------------------------------------------
// Kernel 1: fused QKV projection.  out16 = (f16) relu(X @ W + b); z==0 also
// keeps f32 q for the residual.  Tile 128x64, K-step 32; 8 waves, each 32x32.
// ---------------------------------------------------------------------------
__global__ __launch_bounds__(256) void qkv_proj_wmma(
    const float* __restrict__ Xq, const float* __restrict__ Xk,
    const float* __restrict__ Xv,
    const float* __restrict__ Wq, const float* __restrict__ Wk,
    const float* __restrict__ Wv,
    const float* __restrict__ bq, const float* __restrict__ bk,
    const float* __restrict__ bv,
    _Float16* __restrict__ q16, _Float16* __restrict__ k16,
    _Float16* __restrict__ v16, float* __restrict__ qf32) {
  const int z = blockIdx.z;
  const float* X    = (z == 0) ? Xq : (z == 1) ? Xk : Xv;
  const float* W    = (z == 0) ? Wq : (z == 1) ? Wk : Wv;
  const float* bias = (z == 0) ? bq : (z == 1) ? bk : bv;
  _Float16* out16   = (z == 0) ? q16 : (z == 1) ? k16 : v16;

  const int m0   = blockIdx.x * 128;
  const int n0   = blockIdx.y * 64;
  const int tid  = threadIdx.x;
  const int lane = tid & 31;
  const int wid  = tid >> 5;
  const int wm   = (wid & 3) * 32;
  const int wn   = (wid >> 2) * 32;
  const int rsel = (lane >> 4) * 8;

  __shared__ _Float16 shA[128][40];  // [M][K] +8 pad
  __shared__ _Float16 shB[64][40];   // [N][K] +8 pad

  const v8f zero = {};
  v8f acc[2][2];
#pragma unroll
  for (int i = 0; i < 2; ++i)
#pragma unroll
    for (int j = 0; j < 2; ++j) acc[i][j] = zero;

  for (int kt = 0; kt < EMB; kt += 32) {
    __syncthreads();
    // Phase 1: batch all global loads (back-to-back issue, single wait).
    float4 fa[4], fb[2];
#pragma unroll
    for (int i = 0; i < 4; ++i) {
      int e = tid + i * 256, row = e >> 3, c4 = (e & 7) << 2;
      fa[i] = *(const float4*)(X + (size_t)(m0 + row) * EMB + kt + c4);
    }
#pragma unroll
    for (int i = 0; i < 2; ++i) {
      int e = tid + i * 256, kr = e >> 4, c4 = (e & 15) << 2;
      fb[i] = *(const float4*)(W + (size_t)(kt + kr) * ATT + n0 + c4);
    }
    if (kt + 32 < EMB) {  // -> global_prefetch_b8
      __builtin_prefetch(X + (size_t)(m0 + (tid >> 1)) * EMB + kt + 32, 0, 0);
      __builtin_prefetch(W + (size_t)(kt + 32 + (tid & 31)) * ATT + n0, 0, 0);
    }
    // Phase 2: convert + LDS stores.
#pragma unroll
    for (int i = 0; i < 4; ++i) {
      int e = tid + i * 256, row = e >> 3, c4 = (e & 7) << 2;
      shA[row][c4 + 0] = (_Float16)fa[i].x;
      shA[row][c4 + 1] = (_Float16)fa[i].y;
      shA[row][c4 + 2] = (_Float16)fa[i].z;
      shA[row][c4 + 3] = (_Float16)fa[i].w;
    }
#pragma unroll
    for (int i = 0; i < 2; ++i) {
      int e = tid + i * 256, kr = e >> 4, c4 = (e & 15) << 2;
      shB[c4 + 0][kr] = (_Float16)fb[i].x;
      shB[c4 + 1][kr] = (_Float16)fb[i].y;
      shB[c4 + 2][kr] = (_Float16)fb[i].z;
      shB[c4 + 3][kr] = (_Float16)fb[i].w;
    }
    __syncthreads();

    v16h af[2], bf[2];
#pragma unroll
    for (int mt = 0; mt < 2; ++mt)
      af[mt] = frag_ld(&shA[wm + mt * 16 + (lane & 15)][rsel]);
#pragma unroll
    for (int nt = 0; nt < 2; ++nt)
      bf[nt] = frag_ld(&shB[wn + nt * 16 + (lane & 15)][rsel]);
#pragma unroll
    for (int mt = 0; mt < 2; ++mt)
#pragma unroll
      for (int nt = 0; nt < 2; ++nt)
        acc[mt][nt] = wmma_f16(af[mt], bf[nt], acc[mt][nt]);
  }

#pragma unroll
  for (int mt = 0; mt < 2; ++mt)
#pragma unroll
    for (int nt = 0; nt < 2; ++nt) {
      const float bv_ = bias[n0 + wn + nt * 16 + (lane & 15)];
#pragma unroll
      for (int r = 0; r < 8; ++r) {
        int grow = m0 + wm + mt * 16 + r + rsel;
        int gcol = n0 + wn + nt * 16 + (lane & 15);
        float val = fmaxf(acc[mt][nt][r] + bv_, 0.0f);
        out16[(size_t)grow * ATT + gcol] = (_Float16)val;
        if (z == 0) qf32[(size_t)grow * ATT + gcol] = val;
      }
    }
}

// ---------------------------------------------------------------------------
// Kernel 2: flash attention.  One block per (b, h, 128-query chunk); each of
// 8 waves owns 16 query rows.  Q staged via async global->LDS (ASYNCcnt);
// K tiles DMA'd by the Tensor Data Mover with LDS padding (TENSORcnt);
// V transposed through VGPRs (overlapped with the TDM).  16 WMMAs / K-block.
// ---------------------------------------------------------------------------
__global__ __launch_bounds__(256) void flash_attn_wmma(
    const _Float16* __restrict__ q16, const _Float16* __restrict__ k16,
    const _Float16* __restrict__ vv16, float* __restrict__ attn) {
  const int q0   = blockIdx.x * 128;
  const int b    = blockIdx.y / HEADS;
  const int h    = blockIdx.y % HEADS;
  const int tid  = threadIdx.x;
  const int lane = tid & 31;
  const int wid  = tid >> 5;
  const int rsel = (lane >> 4) * 8;

  __shared__ _Float16 shQ[128][72];    // [q][d]     pitch 144 B
  __shared__ _Float16 shK[64][72];     // [key][d]   (B for QK^T), TDM target
  __shared__ _Float16 shVT[64][72];    // [d][key]   (B for P*V)
  __shared__ _Float16 shP[8][16][72];  // per-wave P relayout (C -> A frag)

  const size_t rowbase = (size_t)b * SEQ;
  const size_t hoff    = (size_t)h * HD;

  // ---- Q chunk: async global->LDS copies, 16B per lane-op (ASYNCcnt). ----
  {
    const uint32_t q_lds = (uint32_t)(uintptr_t)(void*)&shQ[0][0];
#pragma unroll
    for (int i = 0; i < 4; ++i) {
      int e = tid + i * 256;          // 1024 16-byte chunks (pad left alone)
      int row = e >> 3, dc = (e & 7) * 8;
      uint32_t lds = q_lds + (uint32_t)(row * 144 + dc * 2);
      uint64_t ga =
          (uint64_t)(uintptr_t)(q16 + (rowbase + q0 + row) * ATT + hoff + dc);
      asm volatile("global_load_async_to_lds_b128 %0, %1, off" ::"v"(lds),
                   "v"(ga)
                   : "memory");
    }
    asm volatile("s_wait_asynccnt 0x0" ::: "memory");
  }
  __syncthreads();

  v16h qf[2];
#pragma unroll
  for (int kc = 0; kc < 2; ++kc)
    qf[kc] = frag_ld(&shQ[wid * 16 + (lane & 15)][kc * 32 + rsel]);

  const v8f zero = {};
  v8f o[4];
#pragma unroll
  for (int nt = 0; nt < 4; ++nt) o[nt] = zero;
  float m[8], l[8];
#pragma unroll
  for (int r = 0; r < 8; ++r) { m[r] = -3.0e38f; l[r] = 0.0f; }

  const int nkb = q0 / 64 + 2;  // keys 0 .. q0+127 (causal bound)
  for (int kbI = 0; kbI < nkb; ++kbI) {
    const int kb = kbI * 64;
    __syncthreads();  // previous block's K/V reads done before overwrite

    // ---- K tile: Tensor Data Mover DMA, wave 0 issues one descriptor. ----
    // D# group0: count=1, lds_addr, 57b global addr, type=2.
    // D# group1: data_size=2B, pad_enable, pad_interval=32 DW (code 4),
    //            pad_amount=4 DW (code 3) -> 72-half LDS pitch;
    //            tensor_dim0=64, tensor_dim1=64, tile 64x64, stride0=1024.
    if (tid < 32) {
      uint64_t ga =
          (uint64_t)(uintptr_t)(k16 + (rowbase + kb) * ATT + hoff);
      u32x4 g0 = {1u, (uint32_t)(uintptr_t)(void*)&shK[0][0], (uint32_t)ga,
                  (uint32_t)(ga >> 32) | (2u << 30)};
      i32x8 g1 = {(int)((1u << 16) | (1u << 20) | (4u << 22) | (3u << 25)),
                  64 << 16,   // tensor_dim0[15:0] in bits 63:48
                  64 << 16,   // tensor_dim1[15:0] in bits 95:80
                  64 << 16,   // tile_dim0 in bits 127:112
                  64,         // tile_dim1
                  1024,       // tensor_dim0_stride[31:0]
                  0, 0};
      i32x4 gz4 = {0, 0, 0, 0};
      i32x8 gz8 = {0, 0, 0, 0, 0, 0, 0, 0};
      // 6-arg (clang-23 / therock) form: (g0, g1, g2, g3, g4, cpol)
      __builtin_amdgcn_tensor_load_to_lds(g0, g1, gz4, gz4, gz8, 0);
    }

    // ---- V tile: transposed through VGPRs, overlapped with the TDM. ----
#pragma unroll
    for (int i = 0; i < 2; ++i) {
      int e = tid + i * 256;          // 512 chunks of 8 halves
      int key = e >> 3, dc = (e & 7) * 8;
      float4 raw =
          *(const float4*)(vv16 + (rowbase + kb + key) * ATT + hoff + dc);
      const _Float16* hp = (const _Float16*)&raw;
#pragma unroll
      for (int j = 0; j < 8; ++j) shVT[dc + j][key] = hp[j];
    }
    if (tid < 32) __builtin_amdgcn_s_wait_tensorcnt(0);
    __syncthreads();

    // ---- S = Q K^T ----
    v8f sc[4];
#pragma unroll
    for (int nt = 0; nt < 4; ++nt) {
      v8f c = zero;
#pragma unroll
      for (int kc = 0; kc < 2; ++kc) {
        v16h bf = frag_ld(&shK[nt * 16 + (lane & 15)][kc * 32 + rsel]);
        c = wmma_f16(qf[kc], bf, c);
      }
      sc[nt] = c;
    }

    // ---- scale, causal mask, online softmax ----
    float nm[8];
#pragma unroll
    for (int r = 0; r < 8; ++r) nm[r] = m[r];
#pragma unroll
    for (int nt = 0; nt < 4; ++nt) {
      int col = kb + nt * 16 + (lane & 15);
#pragma unroll
      for (int r = 0; r < 8; ++r) {
        int row = q0 + wid * 16 + r + rsel;
        float s = sc[nt][r] * 0.125f;
        s = (col > row) ? NEGF : s;
        sc[nt][r] = s;
        nm[r] = fmaxf(nm[r], s);
      }
    }
#pragma unroll
    for (int r = 0; r < 8; ++r)
#pragma unroll
      for (int off = 1; off < 16; off <<= 1)
        nm[r] = fmaxf(nm[r], __shfl_xor(nm[r], off, 32));

    float rs[8];
#pragma unroll
    for (int r = 0; r < 8; ++r) rs[r] = 0.0f;
#pragma unroll
    for (int nt = 0; nt < 4; ++nt)
#pragma unroll
      for (int r = 0; r < 8; ++r) {
        float p = __expf(sc[nt][r] - nm[r]);
        sc[nt][r] = p;
        rs[r] += p;
      }
#pragma unroll
    for (int r = 0; r < 8; ++r)
#pragma unroll
      for (int off = 1; off < 16; off <<= 1)
        rs[r] += __shfl_xor(rs[r], off, 32);

#pragma unroll
    for (int r = 0; r < 8; ++r) {
      float alpha = __expf(m[r] - nm[r]);
      l[r] = l[r] * alpha + rs[r];
      m[r] = nm[r];
#pragma unroll
      for (int nt = 0; nt < 4; ++nt) o[nt][r] *= alpha;
    }

    // ---- P relayout (C -> A fragment) via wave-private LDS strip ----
#pragma unroll
    for (int nt = 0; nt < 4; ++nt)
#pragma unroll
      for (int r = 0; r < 8; ++r)
        shP[wid][r + rsel][nt * 16 + (lane & 15)] = (_Float16)sc[nt][r];

    v16h pf[2];
#pragma unroll
    for (int kc = 0; kc < 2; ++kc)
      pf[kc] = frag_ld(&shP[wid][lane & 15][kc * 32 + rsel]);

    // ---- O += P V ----
#pragma unroll
    for (int nt = 0; nt < 4; ++nt)
#pragma unroll
      for (int kc = 0; kc < 2; ++kc) {
        v16h bf = frag_ld(&shVT[nt * 16 + (lane & 15)][kc * 32 + rsel]);
        o[nt] = wmma_f16(pf[kc], bf, o[nt]);
      }
  }

  // ---- normalize and write the head's slice (f32) ----
#pragma unroll
  for (int r = 0; r < 8; ++r) {
    float inv = 1.0f / l[r];
    size_t grow = rowbase + q0 + wid * 16 + r + rsel;
#pragma unroll
    for (int nt = 0; nt < 4; ++nt)
      attn[grow * ATT + hoff + nt * 16 + (lane & 15)] = o[nt][r] * inv;
  }
}

// ---------------------------------------------------------------------------
// Kernel 3: residual (+ projected q) and LayerNorm over the 1024-wide row.
// ---------------------------------------------------------------------------
__global__ __launch_bounds__(256) void resid_layernorm(
    const float* __restrict__ attn, const float* __restrict__ qres,
    const float* __restrict__ gamma, const float* __restrict__ beta,
    float* __restrict__ out) {
  const int row = blockIdx.x;
  const int tid = threadIdx.x;
  const size_t base = (size_t)row * ATT + tid * 4;

  float4 a = *(const float4*)(attn + base);
  float4 q = *(const float4*)(qres + base);
  float x0 = a.x + q.x, x1 = a.y + q.y, x2 = a.z + q.z, x3 = a.w + q.w;

  __shared__ float r1[256], r2[256];
  r1[tid] = x0 + x1 + x2 + x3;
  r2[tid] = x0 * x0 + x1 * x1 + x2 * x2 + x3 * x3;
  __syncthreads();
#pragma unroll
  for (int off = 128; off > 0; off >>= 1) {
    if (tid < off) { r1[tid] += r1[tid + off]; r2[tid] += r2[tid + off]; }
    __syncthreads();
  }
  const float mu   = r1[0] * (1.0f / ATT);
  const float var  = r2[0] * (1.0f / ATT) - mu * mu;
  const float rinv = rsqrtf(var + 1e-5f);

  float4 g  = *(const float4*)(gamma + tid * 4);
  float4 be = *(const float4*)(beta + tid * 4);
  float4 y;
  y.x = (x0 - mu) * rinv * g.x + be.x;
  y.y = (x1 - mu) * rinv * g.y + be.y;
  y.z = (x2 - mu) * rinv * g.z + be.z;
  y.w = (x3 - mu) * rinv * g.w + be.w;
  *(float4*)(out + base) = y;
}

// ---------------------------------------------------------------------------
extern "C" void kernel_launch(void* const* d_in, const int* in_sizes, int n_in,
                              void* d_out, int out_size, void* d_ws,
                              size_t ws_size, hipStream_t stream) {
  const float* query = (const float*)d_in[0];
  const float* key   = (const float*)d_in[1];
  const float* value = (const float*)d_in[2];
  const float* Wq    = (const float*)d_in[3];
  const float* bq    = (const float*)d_in[4];
  const float* Wk    = (const float*)d_in[5];
  const float* bk    = (const float*)d_in[6];
  const float* Wv    = (const float*)d_in[7];
  const float* bv    = (const float*)d_in[8];
  const float* gamma = (const float*)d_in[9];
  const float* beta  = (const float*)d_in[10];

  // Workspace: q16/k16/v16 (8 MiB each) | q f32 (16 MiB) | attn f32 (16 MiB)
  char* ws = (char*)d_ws;
  const size_t F16B = (size_t)MTOT * ATT * sizeof(_Float16);
  const size_t F32B = (size_t)MTOT * ATT * sizeof(float);
  _Float16* q16  = (_Float16*)(ws);
  _Float16* k16  = (_Float16*)(ws + F16B);
  _Float16* v16  = (_Float16*)(ws + 2 * F16B);
  float*    qf32 = (float*)(ws + 3 * F16B);
  float*    attn = (float*)(ws + 3 * F16B + F32B);

  dim3 gProj(MTOT / 128, ATT / 64, 3);
  qkv_proj_wmma<<<gProj, 256, 0, stream>>>(query, key, value, Wq, Wk, Wv, bq,
                                           bk, bv, q16, k16, v16, qf32);

  dim3 gAttn(SEQ / 128, BATCH * HEADS);
  flash_attn_wmma<<<gAttn, 256, 0, stream>>>(q16, k16, v16, attn);

  resid_layernorm<<<MTOT, 256, 0, stream>>>(attn, qf32, gamma, beta,
                                            (float*)d_out);
}
